// Trans2DSubNet_33139967656345
// MI455X (gfx1250) — matrix operand order repaired and checked by reading, not compiled, not run on hardware
//
#include <hip/hip_runtime.h>
#include <hip/hip_bf16.h>

typedef __attribute__((ext_vector_type(16))) _Float16 v16h;
typedef __attribute__((ext_vector_type(8)))  float    v8f;
typedef unsigned int v4u __attribute__((ext_vector_type(4)));
typedef unsigned int v8u __attribute__((ext_vector_type(8)));

__device__ inline v8f v8f_zero() {
  v8f z;
#pragma unroll
  for (int i = 0; i < 8; ++i) z[i] = 0.f;
  return z;
}

// A-operand (16x32 f16) per-lane K index: CDNA5 ISA "16-bit A-Matrix 16x32"
__device__ inline int amap(int e, int khalf) {
  int vg = e >> 1, pos = e & 1;
  return 16 * (vg >> 2) + 2 * (vg & 3) + 8 * khalf + pos;
}
// B-operand (32x16 f16) per-lane K index
__device__ inline int bmap(int e, int khalf) {
  int vg = e >> 1, pos = e & 1;
  return 16 * (vg >> 2) + 2 * (vg & 3) + 8 * khalf + pos;
}

// Tensor Data Mover: 2D tile load Global -> LDS, f32 elements.
// D# per CDNA5 ISA (group0 128b + group1 256b).
__device__ inline void tdm_load_2d_f32(unsigned lds_off, const void* gaddr,
                                       unsigned tensor_w, unsigned tensor_h,
                                       unsigned long long row_stride_elems,
                                       unsigned tile_w, unsigned tile_h)
{
  unsigned long long ga = (unsigned long long)gaddr;
  v4u g0;
  g0[0] = 1u;                                       // count=1, user descriptor
  g0[1] = lds_off;                                  // lds_addr (bytes)
  g0[2] = (unsigned)(ga & 0xffffffffu);             // global_addr[31:0]
  g0[3] = (unsigned)((ga >> 32) & 0x01ffffffu)      // global_addr[56:32]
        | (2u << 30);                               // type = 2 ("image")
  v8u g1;
  g1[0] = (2u << 16);                               // wg_mask=0, data_size=4B
  g1[1] = (tensor_w & 0xffffu) << 16;               // tensor_dim0[15:0]
  g1[2] = (tensor_w >> 16) | ((tensor_h & 0xffffu) << 16);   // dim0 hi | dim1 lo
  g1[3] = (tensor_h >> 16) | (tile_w << 16);        // dim1 hi | tile_dim0
  g1[4] = tile_h & 0xffffu;                         // tile_dim1 | tile_dim2=0
  g1[5] = (unsigned)(row_stride_elems & 0xffffffffu);        // dim0_stride lo
  g1[6] = (unsigned)((row_stride_elems >> 32) & 0xffffu);    // dim0_stride hi
  g1[7] = 0u;                                       // dim1_stride (unused, 2D)
  asm volatile("tensor_load_to_lds %0, %1" :: "s"(g0), "s"(g1) : "memory");
}

#define BK 32

// C[M,N] = act(A[M,K] @ W[N,K]^T + bias [+ res]) ; M%128==0, N%64==0, K%32==0
// Double-buffered: TDM for tile k+1 is in flight during WMMA on tile k.
__global__ void __launch_bounds__(256)
gemm_wmma(const float* __restrict__ A, const float* __restrict__ W,
          const float* __restrict__ bias, const float* __restrict__ res,
          float* __restrict__ C, int M, int N, int K, int relu)
{
  __shared__ float Asf[2][128][BK];   // 2 x 16 KB, TDM destinations (contiguous tiles)
  __shared__ float Wsf[2][64][BK];    // 2 x  8 KB
  const int tid  = threadIdx.x;
  const int wave = tid >> 5;
  const int lane = tid & 31;
  const int l16  = lane & 15;
  const int kh   = lane >> 4;
  const int m0 = blockIdx.y * 128;
  const int n0 = blockIdx.x * 64;

  v8f acc[4];
#pragma unroll
  for (int i = 0; i < 4; ++i) acc[i] = v8f_zero();

  // preload tile 0 into buffer 0 (TDM issue is per-wave, EXEC-independent)
  if (wave == 0) {
    tdm_load_2d_f32((unsigned)(uintptr_t)&Asf[0][0][0], A + (size_t)m0 * K,
                    (unsigned)K, (unsigned)M, (unsigned long long)K, BK, 128);
    tdm_load_2d_f32((unsigned)(uintptr_t)&Wsf[0][0][0], W + (size_t)n0 * K,
                    (unsigned)K, (unsigned)N, (unsigned long long)K, BK, 64);
    __builtin_amdgcn_s_wait_tensorcnt(0);
  }
  __syncthreads();

  int cur = 0;
  for (int k0 = 0; k0 < K; k0 += BK) {
    const bool more = (k0 + BK) < K;
    if (wave == 0 && more) {           // kick DMA of next tile into alternate buffer
      int nb = cur ^ 1;
      tdm_load_2d_f32((unsigned)(uintptr_t)&Asf[nb][0][0],
                      A + (size_t)m0 * K + k0 + BK,
                      (unsigned)K, (unsigned)M, (unsigned long long)K, BK, 128);
      tdm_load_2d_f32((unsigned)(uintptr_t)&Wsf[nb][0][0],
                      W + (size_t)n0 * K + k0 + BK,
                      (unsigned)K, (unsigned)N, (unsigned long long)K, BK, 64);
    }

    v16h a;                            // fragment build: ds_load_b128 + cvt (co-exec w/ WMMA)
#pragma unroll
    for (int e = 0; e < 16; ++e) a[e] = (_Float16)Asf[cur][wave * 16 + l16][amap(e, kh)];
#pragma unroll
    for (int nt = 0; nt < 4; ++nt) {
      v16h b;
#pragma unroll
      for (int e = 0; e < 16; ++e) b[e] = (_Float16)Wsf[cur][nt * 16 + l16][bmap(e, kh)];
      acc[nt] = __builtin_amdgcn_wmma_f32_16x16x32_f16(
          false, a, false, b, (short)0, acc[nt], false, false);
    }

    if (wave == 0 && more) __builtin_amdgcn_s_wait_tensorcnt(0);  // next tile landed
    __syncthreads();                   // everyone sees it; old buffer reads complete
    cur ^= 1;
  }

#pragma unroll
  for (int nt = 0; nt < 4; ++nt) {
#pragma unroll
    for (int r = 0; r < 8; ++r) {
      int m = m0 + wave * 16 + r + 8 * kh;  // C layout: VGPR r, lane half selects M
      int n = n0 + nt * 16 + l16;
      float v = acc[nt][r] + bias[n];
      if (res)  v += res[(size_t)m * N + n];
      if (relu) v = fmaxf(v, 0.f);
      C[(size_t)m * N + n] = v;
    }
  }
}

// one block per (batch, head); qkv rows: [q(512) | k(512) | v(512)]
__global__ void __launch_bounds__(128)
attn_wmma(const float* __restrict__ qkv, float* __restrict__ out)
{
  __shared__ _Float16 Kt[128][72];
  __shared__ _Float16 Vt[128][72];
  __shared__ _Float16 Pst[4][16][136];
  const int b = blockIdx.x >> 3;
  const int h = blockIdx.x & 7;
  const int tid  = threadIdx.x;
  const int wave = tid >> 5;
  const int lane = tid & 31;
  const int l16  = lane & 15;
  const int kh   = lane >> 4;

  {
    const float* kg = qkv + (size_t)(b * 128 + tid) * 1536 + 512 + h * 64;
    const float* vg = kg + 512;
#pragma unroll 8
    for (int d = 0; d < 64; ++d) {
      Kt[tid][d] = (_Float16)kg[d];
      Vt[tid][d] = (_Float16)vg[d];
    }
  }
  __syncthreads();

  for (int pass = 0; pass < 2; ++pass) {
    const int r0 = (wave + pass * 4) * 16;
    // Q fragments straight from global (f32 -> f16)
    v16h aq0, aq1;
    const float* qg = qkv + (size_t)(b * 128 + r0 + l16) * 1536 + h * 64;
#pragma unroll
    for (int e = 0; e < 16; ++e) {
      aq0[e] = (_Float16)qg[amap(e, kh)];
      aq1[e] = (_Float16)qg[32 + amap(e, kh)];
    }
    // scores = Q @ K^T  (16 x 128)
    v8f sc[8];
#pragma unroll
    for (int nt = 0; nt < 8; ++nt) {
      v16h b0, b1;
#pragma unroll
      for (int e = 0; e < 16; ++e) {
        b0[e] = Kt[nt * 16 + l16][bmap(e, kh)];
        b1[e] = Kt[nt * 16 + l16][32 + bmap(e, kh)];
      }
      v8f s = v8f_zero();
      s = __builtin_amdgcn_wmma_f32_16x16x32_f16(false, aq0, false, b0, (short)0, s, false, false);
      s = __builtin_amdgcn_wmma_f32_16x16x32_f16(false, aq1, false, b1, (short)0, s, false, false);
      sc[nt] = s;
    }
    // softmax over 128 keys: row m lives in one VGPR across 16 lanes of a half-wave
#pragma unroll
    for (int r = 0; r < 8; ++r) {
      float mx = -1e30f;
#pragma unroll
      for (int nt = 0; nt < 8; ++nt) mx = fmaxf(mx, sc[nt][r]);
      for (int d = 1; d < 16; d <<= 1) mx = fmaxf(mx, __shfl_xor(mx, d, 32));
      float sm = 0.f;
#pragma unroll
      for (int nt = 0; nt < 8; ++nt) {
        float p = __expf(0.125f * (sc[nt][r] - mx));   // 1/sqrt(64) scale
        sc[nt][r] = p;
        sm += p;
      }
      for (int d = 1; d < 16; d <<= 1) sm += __shfl_xor(sm, d, 32);
      float inv = 1.f / sm;
      int m = r + 8 * kh;
#pragma unroll
      for (int nt = 0; nt < 8; ++nt)
        Pst[wave][m][nt * 16 + l16] = (_Float16)(sc[nt][r] * inv);
    }
    // out = P @ V  (16x128 @ 128x64); P restaged in LDS as A-operand
    v8f ov[4];
#pragma unroll
    for (int i = 0; i < 4; ++i) ov[i] = v8f_zero();
#pragma unroll
    for (int kk = 0; kk < 128; kk += 32) {
      v16h ap;
#pragma unroll
      for (int e = 0; e < 16; ++e) ap[e] = Pst[wave][l16][kk + amap(e, kh)];
#pragma unroll
      for (int nt = 0; nt < 4; ++nt) {
        v16h bv;
#pragma unroll
        for (int e = 0; e < 16; ++e) bv[e] = Vt[kk + bmap(e, kh)][nt * 16 + l16];
        ov[nt] = __builtin_amdgcn_wmma_f32_16x16x32_f16(
            false, ap, false, bv, (short)0, ov[nt], false, false);
      }
    }
#pragma unroll
    for (int nt = 0; nt < 4; ++nt)
#pragma unroll
      for (int r = 0; r < 8; ++r) {
        int m = r0 + r + 8 * kh;
        out[(size_t)(b * 128 + m) * 512 + h * 64 + nt * 16 + l16] = ov[nt][r];
      }
  }
}

__global__ void __launch_bounds__(256)
layernorm_inplace(float* __restrict__ x, const float* __restrict__ g,
                  const float* __restrict__ bt, int D)
{
  __shared__ float red[256];
  float* xr = x + (size_t)blockIdx.x * D;
  const int tid = threadIdx.x;
  float s = 0.f;
  for (int i = tid; i < D; i += 256) s += xr[i];
  red[tid] = s; __syncthreads();
  for (int o = 128; o > 0; o >>= 1) { if (tid < o) red[tid] += red[tid + o]; __syncthreads(); }
  float mu = red[0] / D;
  __syncthreads();
  float v = 0.f;
  for (int i = tid; i < D; i += 256) { float d = xr[i] - mu; v += d * d; }
  red[tid] = v; __syncthreads();
  for (int o = 128; o > 0; o >>= 1) { if (tid < o) red[tid] += red[tid + o]; __syncthreads(); }
  float rstd = rsqrtf(red[0] / D + 1e-5f);
  for (int i = tid; i < D; i += 256) xr[i] = (xr[i] - mu) * rstd * g[i] + bt[i];
}

__global__ void __launch_bounds__(256)
seq_mean(const float* __restrict__ x, float* __restrict__ pooled)
{
  int b = blockIdx.x;
  for (int d = threadIdx.x; d < 512; d += 256) {
    float s = 0.f;
    for (int t = 0; t < 128; ++t) s += x[(size_t)(b * 128 + t) * 512 + d];
    pooled[b * 512 + d] = s * (1.f / 128.f);
  }
}

// ---------- complex scalar helpers for root-MUSIC ----------
struct c32 { float x, y; };
__device__ inline c32 mkc(float a, float b) { c32 r; r.x = a; r.y = b; return r; }
__device__ inline c32 cadd(c32 a, c32 b) { return mkc(a.x + b.x, a.y + b.y); }
__device__ inline c32 csub(c32 a, c32 b) { return mkc(a.x - b.x, a.y - b.y); }
__device__ inline c32 cmul(c32 a, c32 b) { return mkc(a.x * b.x - a.y * b.y, a.x * b.y + a.y * b.x); }
__device__ inline c32 conjg(c32 a) { return mkc(a.x, -a.y); }
__device__ inline c32 cscale(c32 a, float s) { return mkc(a.x * s, a.y * s); }
__device__ inline float cabs2(c32 a) { return a.x * a.x + a.y * a.y; }
__device__ inline c32 cdiv(c32 a, c32 b) {
  float d = cabs2(b) + 1e-30f;
  return mkc((a.x * b.x + a.y * b.y) / d, (a.y * b.x - a.x * b.y) / d);
}

// one thread per (batch, axis): 8x8 Hermitian Jacobi eigendecomp + Durand-Kerner roots
__global__ void doa_kernel(const float* __restrict__ Rx, const float* __restrict__ Ry,
                           float* __restrict__ doa)
{
  int t = blockIdx.x * blockDim.x + threadIdx.x;
  if (t >= 512) return;
  int b = t >> 1, axis = t & 1;
  const float* R = (axis ? Ry : Rx) + (size_t)b * 128;   // (16,8) row-major

  c32 Kc[8][8];
  for (int r = 0; r < 8; ++r)
    for (int c = 0; c < 8; ++c) Kc[r][c] = mkc(R[r * 8 + c], R[(r + 8) * 8 + c]);

  c32 A[8][8];
  for (int i = 0; i < 8; ++i)
    for (int j = 0; j < 8; ++j) {
      c32 s = mkc(i == j ? 1.f : 0.f, 0.f);
      for (int r = 0; r < 8; ++r) s = cadd(s, cmul(conjg(Kc[r][i]), Kc[r][j]));
      A[i][j] = s;
    }

  c32 V[8][8];
  for (int i = 0; i < 8; ++i)
    for (int j = 0; j < 8; ++j) V[i][j] = mkc(i == j ? 1.f : 0.f, 0.f);

  for (int sweep = 0; sweep < 12; ++sweep)
    for (int p = 0; p < 7; ++p)
      for (int q = p + 1; q < 8; ++q) {
        c32 apq = A[p][q];
        float nrm = sqrtf(cabs2(apq));
        if (nrm < 1e-12f) continue;
        c32 ph = cscale(apq, 1.f / nrm);
        float tau = (A[q][q].x - A[p][p].x) / (2.f * nrm);
        float tt = (tau >= 0.f ? 1.f : -1.f) / (fabsf(tau) + sqrtf(1.f + tau * tau));
        float c_ = rsqrtf(1.f + tt * tt);
        float s_ = tt * c_;
        c32 sp = cscale(ph, s_);
        for (int k = 0; k < 8; ++k) {       // right-multiply (columns) on A and V
          c32 akp = A[k][p], akq = A[k][q];
          A[k][p] = csub(cscale(akp, c_), cmul(conjg(sp), akq));
          A[k][q] = cadd(cmul(sp, akp), cscale(akq, c_));
          c32 vkp = V[k][p], vkq = V[k][q];
          V[k][p] = csub(cscale(vkp, c_), cmul(conjg(sp), vkq));
          V[k][q] = cadd(cmul(sp, vkp), cscale(vkq, c_));
        }
        for (int k = 0; k < 8; ++k) {       // left-multiply by J^H (rows of A)
          c32 apk = A[p][k], aqk = A[q][k];
          A[p][k] = csub(cscale(apk, c_), cmul(sp, aqk));
          A[q][k] = cadd(cmul(conjg(sp), apk), cscale(aqk, c_));
        }
      }

  float ev[8];
  for (int i = 0; i < 8; ++i) ev[i] = A[i][i].x;
  for (int i = 0; i < 7; ++i) {             // ascending eigen-order
    int mi = i;
    for (int j = i + 1; j < 8; ++j) if (ev[j] < ev[mi]) mi = j;
    if (mi != i) {
      float te = ev[i]; ev[i] = ev[mi]; ev[mi] = te;
      for (int k = 0; k < 8; ++k) { c32 tv = V[k][i]; V[k][i] = V[k][mi]; V[k][mi] = tv; }
    }
  }

  c32 F[8][8];                              // noise projector Un Un^H, Un = 5 smallest
  for (int i = 0; i < 8; ++i)
    for (int j = 0; j < 8; ++j) {
      c32 s = mkc(0.f, 0.f);
      for (int k = 0; k < 5; ++k) s = cadd(s, cmul(V[i][k], conjg(V[j][k])));
      F[i][j] = s;
    }

  c32 cf[15];                               // diagonal sums, o = -7..7
  for (int o = -7; o <= 7; ++o) {
    c32 s = mkc(0.f, 0.f);
    for (int i = 0; i < 8; ++i) {
      int j = i + o;
      if (j >= 0 && j < 8) s = cadd(s, F[i][j]);
    }
    cf[o + 7] = s;
  }

  c32 bc[15];
  bc[0] = mkc(1.f, 0.f);
  for (int j = 1; j < 15; ++j) bc[j] = cdiv(cf[j], cf[0]);

  c32 z[14];                                // Durand-Kerner on the degree-14 monic poly
  c32 w = mkc(0.4f, 0.9f);
  z[0] = w;
  for (int k = 1; k < 14; ++k) z[k] = cmul(z[k - 1], w);
  for (int it = 0; it < 64; ++it)
    for (int i = 0; i < 14; ++i) {
      c32 p = bc[0];
      for (int j = 1; j < 15; ++j) p = cadd(cmul(p, z[i]), bc[j]);
      c32 den = mkc(1.f, 0.f);
      for (int j = 0; j < 14; ++j) if (j != i) den = cmul(den, csub(z[i], z[j]));
      z[i] = csub(z[i], cdiv(p, den));
    }

  float key[14]; bool used[14];
  for (int i = 0; i < 14; ++i) {
    float mag = sqrtf(cabs2(z[i]));
    key[i] = fabsf(mag - 1.f) + (mag < 1.f ? 0.f : 1e6f);
    used[i] = false;
  }
  const float INV_PI = 0.318309886183790671f;
  for (int s = 0; s < 3; ++s) {
    int mi = 0; float mk = 1e30f;
    for (int i = 0; i < 14; ++i) if (!used[i] && key[i] < mk) { mk = key[i]; mi = i; }
    used[mi] = true;
    float ang = atan2f(z[mi].y, z[mi].x);
    doa[(size_t)b * 6 + axis * 3 + s] = asinf(ang * INV_PI);
  }
}

__global__ void __launch_bounds__(256)
mlp_kernel(const float* __restrict__ doa,
           const float* __restrict__ w1, const float* __restrict__ b1,
           const float* __restrict__ w2, const float* __restrict__ b2,
           const float* __restrict__ w3, const float* __restrict__ b3,
           float* __restrict__ outp)
{
  __shared__ float h0[8], h1[256], h2[256];
  int b = blockIdx.x, t = threadIdx.x;
  if (t < 6) h0[t] = doa[b * 6 + t];
  __syncthreads();
  { float s = b1[t]; for (int k = 0; k < 6; ++k) s += w1[t * 6 + k] * h0[k]; h1[t] = fmaxf(s, 0.f); }
  __syncthreads();
  { float s = b2[t]; for (int k = 0; k < 256; ++k) s += w2[t * 256 + k] * h1[k]; h2[t] = fmaxf(s, 0.f); }
  __syncthreads();
  if (t < 6) {
    float s = b3[t];
    for (int k = 0; k < 256; ++k) s += w3[t * 256 + k] * h2[k];
    if (t < 3) outp[b * 3 + t] = s;               // outs[:, :M]
    else       outp[768 + b * 3 + (t - 3)] = s;   // outs[:, M:]
  }
}

extern "C" void kernel_launch(void* const* d_in, const int* in_sizes, int n_in,
                              void* d_out, int out_size, void* d_ws, size_t ws_size,
                              hipStream_t stream)
{
  const float* x0   = (const float*)d_in[0];   // (32768, 512) view of Rx_tau
  const float* wqkv = (const float*)d_in[1];
  const float* bqkv = (const float*)d_in[2];
  const float* wo   = (const float*)d_in[3];
  const float* bo   = (const float*)d_in[4];
  const float* g1   = (const float*)d_in[5];
  const float* be1  = (const float*)d_in[6];
  const float* w1   = (const float*)d_in[7];
  const float* b1   = (const float*)d_in[8];
  const float* w2   = (const float*)d_in[9];
  const float* b2   = (const float*)d_in[10];
  const float* g2   = (const float*)d_in[11];
  const float* be2  = (const float*)d_in[12];
  const float* fxw  = (const float*)d_in[13];
  const float* fxb  = (const float*)d_in[14];
  const float* fyw  = (const float*)d_in[15];
  const float* fyb  = (const float*)d_in[16];
  const float* mw1  = (const float*)d_in[17];
  const float* mb1  = (const float*)d_in[18];
  const float* mw2  = (const float*)d_in[19];
  const float* mb2  = (const float*)d_in[20];
  const float* mw3  = (const float*)d_in[21];
  const float* mb3  = (const float*)d_in[22];

  const int Mrows = 256 * 128;                 // 32768
  char* ws = (char*)d_ws;
  size_t off = 0;
  auto alloc = [&](size_t nfloats) {
    float* p = (float*)(ws + off);
    off += nfloats * sizeof(float);
    return p;
  };
  float* bufA   = alloc((size_t)Mrows * 2048); // qkv (uses 1536/row), then ff1 (2048/row)
  float* bufB   = alloc((size_t)Mrows * 512);  // attn_out -> pre2 -> x2
  float* bufC   = alloc((size_t)Mrows * 512);  // x1
  float* pooled = alloc(256 * 512);
  float* Rxb    = alloc(256 * 128);
  float* Ryb    = alloc(256 * 128);
  float* doab   = alloc(256 * 6);

  dim3 blk(256);
  // qkv = x @ Wqkv^T + b
  gemm_wmma<<<dim3(1536 / 64, Mrows / 128), blk, 0, stream>>>(x0, wqkv, bqkv, nullptr, bufA, Mrows, 1536, 512, 0);
  // multi-head attention
  attn_wmma<<<dim3(256 * 8), dim3(128), 0, stream>>>(bufA, bufB);
  // x1 = LN1(attn @ Wo^T + bo + x)
  gemm_wmma<<<dim3(512 / 64, Mrows / 128), blk, 0, stream>>>(bufB, wo, bo, x0, bufC, Mrows, 512, 512, 0);
  layernorm_inplace<<<dim3(Mrows), blk, 0, stream>>>(bufC, g1, be1, 512);
  // ff1 = relu(x1 @ W1^T + b1)
  gemm_wmma<<<dim3(2048 / 64, Mrows / 128), blk, 0, stream>>>(bufC, w1, b1, nullptr, bufA, Mrows, 2048, 512, 1);
  // x2 = LN2(ff1 @ W2^T + b2 + x1)
  gemm_wmma<<<dim3(512 / 64, Mrows / 128), blk, 0, stream>>>(bufA, w2, b2, bufC, bufB, Mrows, 512, 2048, 0);
  layernorm_inplace<<<dim3(Mrows), blk, 0, stream>>>(bufB, g2, be2, 512);
  // pooled mean over sequence
  seq_mean<<<dim3(256), blk, 0, stream>>>(bufB, pooled);
  // fcx / fcy heads (M=256, N=128, K=512)
  gemm_wmma<<<dim3(128 / 64, 256 / 128), blk, 0, stream>>>(pooled, fxw, fxb, nullptr, Rxb, 256, 128, 512, 0);
  gemm_wmma<<<dim3(128 / 64, 256 / 128), blk, 0, stream>>>(pooled, fyw, fyb, nullptr, Ryb, 256, 128, 512, 0);
  // root-MUSIC DOA (eigh + poly roots), 512 independent problems
  doa_kernel<<<dim3(8), dim3(64), 0, stream>>>(Rxb, Ryb, doab);
  // final MLP -> d_out (two (256,3) outputs concatenated)
  mlp_kernel<<<dim3(256), blk, 0, stream>>>(doab, mw1, mb1, mw2, mb2, mw3, mb3, (float*)d_out);

  (void)in_sizes; (void)n_in; (void)out_size; (void)ws_size;
}
